// X_Agent_3272765079558
// MI455X (gfx1250) — matrix operand order repaired and checked by reading, not compiled, not run on hardware
//
#include <hip/hip_runtime.h>

// ---------------- problem constants ----------------
#define NUM_LAYERS 12
#define AGENT_LEN  10
#define EMBED      768
#define TEXT_DIM   512
#define N_TOK      1024
#define BATCH      32
#define N_CLS      150
#define N_CTX      77
#define SCALE_D    0.036084391824351615f  /* 768^-0.5 */

typedef __attribute__((ext_vector_type(16))) __bf16 v16bf;
typedef __attribute__((ext_vector_type(8)))  float  v8f;
typedef __attribute__((ext_vector_type(4)))  int    v4i;

// CDNA5 async global->LDS DMA (ASYNCcnt-tracked), guarded for toolchain drift.
#if defined(__has_builtin)
#  if __has_builtin(__builtin_amdgcn_global_load_async_to_lds_b128) && \
      __has_builtin(__builtin_amdgcn_s_wait_asynccnt)
#    define HAVE_ASYNC_LDS 1
#  endif
#endif
#ifndef HAVE_ASYNC_LDS
#  define HAVE_ASYNC_LDS 0
#endif

// per the compile probe: param0 is int4 in AS(1) ("__device__"), param1 LDS side
#define AS1_V4I(p) ((__attribute__((address_space(1))) v4i*)(p))
#define AS3_V4I(p) ((__attribute__((address_space(3))) v4i*)(p))

// ---------------- workspace layout (floats) ----------------
// padded M to multiples of 16: 150->160, 10->16
#define OFF_TFN     0                         // 160 x 512
#define OFF_TEMB    (OFF_TFN    + 160*512)    // 160 x 768
#define OFF_AGP     (OFF_TEMB   + 160*768)    // 16  x 768
#define OFF_MSC     (OFF_AGP    + 16*768)     // 160 x 16   (text-mask scores, pre-sigmoid)
#define OFF_MTPOOL  (OFF_MSC    + 160*16)     // 16  x 768
#define OFF_MT      (OFF_MTPOOL + 16*768)     // 16  x 768  (masked_text final)
#define OFF_VPOOL   (OFF_MT     + 16*768)     // 32 x 16 x 768
#define OFF_VWSUM   (OFF_VPOOL  + 32*16*768)  // 32 x 16
#define OFF_AG2     (OFF_VWSUM  + 32*16)      // 32 x 16 x 768
#define OFF_VPRE    (OFF_AG2    + 32*16*768)  // 32 x 16 x 768
#define OFF_V1      (OFF_VPRE   + 32*16*768)  // 32 x 16 x 768
#define OFF_U       (OFF_V1     + 32*16*768)  // 32 x 16 x 768
#define WS_FLOATS   (OFF_U      + 32*16*768)

// ---------------- helpers ----------------
__global__ void k_zero(float* __restrict__ ws, int n) {
  for (int i = blockIdx.x * 256 + threadIdx.x; i < n; i += gridDim.x * 256)
    ws[i] = 0.0f;
}

__global__ void k_agp(const float* __restrict__ agent, const int* __restrict__ layerp,
                      float* __restrict__ agp) {
  int i = blockIdx.x * 256 + threadIdx.x;          // over 16*768
  if (i >= 16 * EMBED) return;
  int a = i / EMBED, d = i - a * EMBED;
  int L = *layerp;
  agp[i] = (a < AGENT_LEN) ? agent[((size_t)L * AGENT_LEN + a) * EMBED + d] : 0.0f;
}

// tf mean over ctx + L2 normalize -> tfn (150 x 512), pad rows stay zero
__global__ __launch_bounds__(256) void k_textnorm(const float* __restrict__ tf,
                                                  float* __restrict__ tfn) {
  const int c = blockIdx.x;
  __shared__ float red[256];
  float m[2];
  float ss = 0.0f;
#pragma unroll
  for (int j = 0; j < 2; ++j) {
    int d = threadIdx.x + j * 256;
    float s = 0.0f;
    for (int t = 0; t < N_CTX; ++t)
      s += tf[((size_t)c * N_CTX + t) * TEXT_DIM + d];
    m[j] = s * (1.0f / N_CTX);
    ss += m[j] * m[j];
  }
  red[threadIdx.x] = ss;
  __syncthreads();
  for (int o = 128; o > 0; o >>= 1) {
    if (threadIdx.x < o) red[threadIdx.x] += red[threadIdx.x + o];
    __syncthreads();
  }
  float invn = rsqrtf(red[0]);
  __syncthreads();
#pragma unroll
  for (int j = 0; j < 2; ++j) {
    int d = threadIdx.x + j * 256;
    tfn[(size_t)c * TEXT_DIM + d] = m[j] * invn;
  }
}

// ------------- generic WMMA GEMM: C = act(A(MxK) @ W(NxK)^T * premul + bias [+extras]) -------------
// one wave per 16x16 C tile; A/W fp32 in global (b128 loads), converted to bf16 fragments.
// extras (optional) are 16xldc tensors indexed by (row & 15).
__global__ __launch_bounds__(32) void k_gemm_bf16(
    const float* __restrict__ A, int lda,
    const float* __restrict__ W, int ldw,
    const float* __restrict__ bias,
    const float* __restrict__ extra0,
    const float* __restrict__ extra1,
    float* __restrict__ C, int ldc,
    int K, float premul, int relu) {
  const int lane = threadIdx.x & 31;
  const int half = lane >> 4;
  const int r16  = lane & 15;
  const int m0 = blockIdx.x * 16;
  const int n0 = blockIdx.y * 16;

  v8f acc = {};
  const float* arow = A + (size_t)(m0 + r16) * lda;
  const float* wrow = W + (size_t)(n0 + r16) * ldw;
  for (int k0 = 0; k0 < K; k0 += 32) {
    v16bf af, bf;
    // A fragment: lane holds row m0+r16, K = 16h + e (16 consecutive) -> four float4
    const float4* pa0 = reinterpret_cast<const float4*>(arow + k0 + (half << 4));
    float4 a0 = pa0[0], a1 = pa0[1], a2 = pa0[2], a3 = pa0[3];
    af[0] = (__bf16)a0.x; af[1] = (__bf16)a0.y; af[2]  = (__bf16)a0.z; af[3]  = (__bf16)a0.w;
    af[4] = (__bf16)a1.x; af[5] = (__bf16)a1.y; af[6]  = (__bf16)a1.z; af[7]  = (__bf16)a1.w;
    af[8] = (__bf16)a2.x; af[9] = (__bf16)a2.y; af[10] = (__bf16)a2.z; af[11] = (__bf16)a2.w;
    af[12] = (__bf16)a3.x; af[13] = (__bf16)a3.y; af[14] = (__bf16)a3.z; af[15] = (__bf16)a3.w;
    // B fragment: lane holds col n0+r16, K = 16h + e (16 consecutive) -> four float4
    const float4* pb = reinterpret_cast<const float4*>(wrow + k0 + (half << 4));
    float4 b0 = pb[0], b1 = pb[1], b2 = pb[2], b3 = pb[3];
    bf[0] = (__bf16)b0.x; bf[1] = (__bf16)b0.y; bf[2]  = (__bf16)b0.z; bf[3]  = (__bf16)b0.w;
    bf[4] = (__bf16)b1.x; bf[5] = (__bf16)b1.y; bf[6]  = (__bf16)b1.z; bf[7]  = (__bf16)b1.w;
    bf[8] = (__bf16)b2.x; bf[9] = (__bf16)b2.y; bf[10] = (__bf16)b2.z; bf[11] = (__bf16)b2.w;
    bf[12] = (__bf16)b3.x; bf[13] = (__bf16)b3.y; bf[14] = (__bf16)b3.z; bf[15] = (__bf16)b3.w;
    acc = __builtin_amdgcn_wmma_f32_16x16x32_bf16(false, af, false, bf,
                                                  (short)0, acc, false, false);
  }
  const int n = n0 + r16;
  float bv = bias ? bias[n] : 0.0f;
#pragma unroll
  for (int r = 0; r < 8; ++r) {
    int m = m0 + r + (half << 3);
    float v = acc[r] * premul + bv;
    if (extra0) v += extra0[(m & 15) * ldc + n];
    if (extra1) v += extra1[(m & 15) * ldc + n];
    if (relu) v = fmaxf(v, 0.0f);
    C[(size_t)m * ldc + n] = v;
  }
}

// masked_text pooling: mtpool[a,d] = sum_c sigmoid(msc[c,a]) * temb[c,d] / (sum_c sigmoid + 1e-6)
__global__ void k_textpool(const float* __restrict__ msc, const float* __restrict__ temb,
                           float* __restrict__ mtpool) {
  int a = blockIdx.x;                       // 0..9
  int d = blockIdx.y * 128 + threadIdx.x;   // 0..767
  float accv = 0.0f, accw = 0.0f;
  for (int c = 0; c < N_CLS; ++c) {
    float w = 1.0f / (1.0f + __expf(-msc[c * 16 + a]));
    accw += w;
    accv += w * temb[(size_t)c * EMBED + d];
  }
  mtpool[a * EMBED + d] = accv / (accw + 1e-6f);
}

// ------------- masked_vis streaming pass -------------
// grid (32, 4): batch b, quarter of tokens (256). 8 waves/block.
__global__ __launch_bounds__(256) void k_vispool(
    const float* __restrict__ out_in,   // (1025,32,768)
    const float* __restrict__ agp,      // 16x768
    float* __restrict__ vpool,          // 32x16x768 accumulators
    float* __restrict__ vwsum) {        // 32x16 accumulators
  const int b = blockIdx.x;
  const int quarter = blockIdx.y;
  const int tid = threadIdx.x;
  const int lane = tid & 31, wv = tid >> 5;
  const int half = lane >> 4, r16 = lane & 15;

  __shared__ __bf16 s_ag[16][EMBED + 8];  // pad to dodge bank conflicts
  __shared__ __bf16 s_F[16][EMBED];
  __shared__ float  s_sig[16][16];

  for (int i = tid; i < 16 * EMBED; i += 256) {
    int a = i / EMBED, d = i - a * EMBED;
    s_ag[a][d] = (__bf16)agp[i];
  }
  v8f acc[6];
#pragma unroll
  for (int t = 0; t < 6; ++t) acc[t] = (v8f){};
  float wacc = 0.0f;

  for (int ch = 0; ch < 16; ++ch) {
    __syncthreads();  // protect s_F/s_sig reuse (and s_ag on first iter)
    const int n0 = quarter * 256 + ch * 16;
    for (int i = tid; i < (16 * EMBED) / 4; i += 256) {
      int rr = (i * 4) / EMBED, d = (i * 4) % EMBED;
      const float* src = out_in + (size_t)(1 + n0 + rr) * (BATCH * EMBED) +
                         (size_t)b * EMBED + d;
      float4 f = *reinterpret_cast<const float4*>(src);
      if (ch + 1 < 16)  // prefetch next chunk (global_prefetch_b8)
        __builtin_prefetch(src + 16 * BATCH * EMBED, 0, 0);
      s_F[rr][d + 0] = (__bf16)f.x;
      s_F[rr][d + 1] = (__bf16)f.y;
      s_F[rr][d + 2] = (__bf16)f.z;
      s_F[rr][d + 3] = (__bf16)f.w;
    }
    __syncthreads();
    {
      int tok = tid >> 4, a = tid & 15;
      float s = 0.0f;
      if (a < AGENT_LEN) {
        float acd = 0.0f;
        for (int d = 0; d < EMBED; ++d)
          acd += (float)s_F[tok][d] * (float)s_ag[a][d];
        s = 1.0f / (1.0f + __expf(-acd * SCALE_D));
      }
      s_sig[tok][a] = s;
    }
    __syncthreads();
    if (tid < 16) {
#pragma unroll
      for (int tok = 0; tok < 16; ++tok) wacc += s_sig[tok][tid];
    }
    // A = sigma^T (16 agents x 16 tokens, zero-padded K to 32)
    v16bf afr;
#pragma unroll
    for (int e = 0; e < 16; ++e) {
      int kk = (e & 7) + (half << 3) + ((e >> 3) << 4);
      afr[e] = (kk < 16) ? (__bf16)s_sig[kk][r16] : (__bf16)0.0f;
    }
#pragma unroll
    for (int t = 0; t < 6; ++t) {
      int nc = wv * 96 + t * 16 + r16;
      v16bf bfr;
#pragma unroll
      for (int e = 0; e < 16; ++e) {
        int kk = (e & 7) + (half << 3) + ((e >> 3) << 4);
        bfr[e] = (kk < 16) ? s_F[kk][nc] : (__bf16)0.0f;
      }
      acc[t] = __builtin_amdgcn_wmma_f32_16x16x32_bf16(false, afr, false, bfr,
                                                       (short)0, acc[t], false, false);
    }
  }
  float* vp = vpool + (size_t)b * 16 * EMBED;
#pragma unroll
  for (int t = 0; t < 6; ++t) {
    int nc = wv * 96 + t * 16 + r16;
#pragma unroll
    for (int r = 0; r < 8; ++r) {
      int a = r + (half << 3);
      atomicAdd(&vp[a * EMBED + nc], acc[t][r]);
    }
  }
  if (tid < 16) atomicAdd(&vwsum[b * 16 + tid], wacc);
}

__global__ void k_visnorm(float* __restrict__ vpool, const float* __restrict__ vwsum) {
  int b = blockIdx.x;
  for (int i = threadIdx.x; i < 16 * EMBED; i += 256) {
    int a = i / EMBED;
    float w = vwsum[b * 16 + a] + 1e-6f;
    vpool[(size_t)b * 16 * EMBED + i] /= w;
  }
}

// attn_t softmax + vpre = p @ temb  (small; VALU). ag2 tile staged via async DMA to LDS.
__global__ __launch_bounds__(256) void k_attnt(const float* __restrict__ ag2,
                                               const float* __restrict__ temb,
                                               float* __restrict__ vpre) {
  const int b = blockIdx.x;
  __shared__ float s_ag2[16 * EMBED];              // 48 KB
  __shared__ float s_s[AGENT_LEN][N_CLS + 10];
  const float* agb = ag2 + (size_t)b * 16 * EMBED;
#if HAVE_ASYNC_LDS
  for (int i = threadIdx.x * 4; i < 16 * EMBED; i += 256 * 4) {
    __builtin_amdgcn_global_load_async_to_lds_b128(
        AS1_V4I(agb + i), AS3_V4I(&s_ag2[i]), 0, 0);
  }
  __builtin_amdgcn_s_wait_asynccnt(0);
#else
  for (int i = threadIdx.x; i < 16 * EMBED; i += 256) s_ag2[i] = agb[i];
#endif
  __syncthreads();
  for (int i = threadIdx.x; i < AGENT_LEN * N_CLS; i += 256) {
    int a = i / N_CLS, c = i - a * N_CLS;
    const float* tr = temb + (size_t)c * EMBED;
    const float* ar = &s_ag2[a * EMBED];
    float s = 0.0f;
    for (int d = 0; d < EMBED; ++d) s += ar[d] * tr[d];
    s_s[a][c] = s * SCALE_D;
  }
  __syncthreads();
  if (threadIdx.x < AGENT_LEN) {
    int a = threadIdx.x;
    float mx = -1e30f;
    for (int c = 0; c < N_CLS; ++c) mx = fmaxf(mx, s_s[a][c]);
    float sum = 0.0f;
    for (int c = 0; c < N_CLS; ++c) { float e = __expf(s_s[a][c] - mx); s_s[a][c] = e; sum += e; }
    float inv = 1.0f / sum;
    for (int c = 0; c < N_CLS; ++c) s_s[a][c] *= inv;
  }
  __syncthreads();
  for (int i = threadIdx.x; i < AGENT_LEN * EMBED; i += 256) {
    int a = i / EMBED, d = i - a * EMBED;
    float s = 0.0f;
    for (int c = 0; c < N_CLS; ++c) s += s_s[a][c] * temb[(size_t)c * EMBED + d];
    vpre[(size_t)b * 16 * EMBED + a * EMBED + d] = s;
  }
}

__global__ void k_cls(const float* __restrict__ out_in, float* __restrict__ d_out) {
  int i = blockIdx.x * 256 + threadIdx.x;
  if (i < BATCH * EMBED) d_out[i] = out_in[i];
}

// ------------- final fused pass: attn_v softmax + delta = P@u, out = x + (delta+b2)*agent_scale ----
__global__ __launch_bounds__(256) void k_final(
    const float* __restrict__ out_in, const float* __restrict__ ag2,
    const float* __restrict__ u, const float* __restrict__ b2,
    const float* __restrict__ ascale_p, float* __restrict__ d_out) {
  const int b = blockIdx.x;
  const int n0 = blockIdx.y * 16;     // token tile
  const int tid = threadIdx.x;
  const int lane = tid & 31, wv = tid >> 5;
  const int half = lane >> 4, r16 = lane & 15;

  __shared__ __bf16 s_ag[16][EMBED + 8];
  __shared__ __bf16 s_F[16][EMBED];
  __shared__ float  s_sc[16][16];
  __shared__ __bf16 s_P[16][32];

  const float ascale = *ascale_p;
  const float* agb = ag2 + (size_t)b * 16 * EMBED;
  for (int i = tid; i < (16 * EMBED) / 4; i += 256) {
    int a = (i * 4) / EMBED, d = (i * 4) % EMBED;
    const float4 g = *reinterpret_cast<const float4*>(agb + a * EMBED + d);
    s_ag[a][d + 0] = (__bf16)g.x; s_ag[a][d + 1] = (__bf16)g.y;
    s_ag[a][d + 2] = (__bf16)g.z; s_ag[a][d + 3] = (__bf16)g.w;
    const float4 f = *reinterpret_cast<const float4*>(
        out_in + (size_t)(1 + n0 + a) * (BATCH * EMBED) + (size_t)b * EMBED + d);
    s_F[a][d + 0] = (__bf16)f.x; s_F[a][d + 1] = (__bf16)f.y;
    s_F[a][d + 2] = (__bf16)f.z; s_F[a][d + 3] = (__bf16)f.w;
  }
  for (int i = tid; i < 16 * 32; i += 256) (&s_P[0][0])[i] = (__bf16)0.0f;
  __syncthreads();
  {
    int tok = tid >> 4, a = tid & 15;
    float s = -1e30f;
    if (a < AGENT_LEN) {
      float acd = 0.0f;
      for (int d = 0; d < EMBED; ++d)
        acd += (float)s_F[tok][d] * (float)s_ag[a][d];
      s = acd * SCALE_D;
    }
    s_sc[tok][a] = s;
  }
  __syncthreads();
  if (tid < 16) {
    int tok = tid;
    float mx = -1e30f;
#pragma unroll
    for (int a = 0; a < AGENT_LEN; ++a) mx = fmaxf(mx, s_sc[tok][a]);
    float e[AGENT_LEN];
    float sum = 0.0f;
#pragma unroll
    for (int a = 0; a < AGENT_LEN; ++a) { e[a] = __expf(s_sc[tok][a] - mx); sum += e[a]; }
    float inv = 1.0f / sum;
#pragma unroll
    for (int a = 0; a < AGENT_LEN; ++a) s_P[tok][a] = (__bf16)(e[a] * inv);
  }
  __syncthreads();
  // delta tile = P(16 tok x K<=10) @ u_b(K x 768): one WMMA per 16-col slice per wave
  v16bf afr;
#pragma unroll
  for (int e2 = 0; e2 < 16; ++e2) {
    int kk = (e2 & 7) + (half << 3) + ((e2 >> 3) << 4);
    afr[e2] = s_P[r16][kk];          // zero for kk >= 10 by construction
  }
  const float* ub = u + (size_t)b * 16 * EMBED;
#pragma unroll
  for (int t = 0; t < 6; ++t) {
    int nc = wv * 96 + t * 16 + r16;
    v16bf bfr;
#pragma unroll
    for (int e2 = 0; e2 < 16; ++e2) {
      int kk = (e2 & 7) + (half << 3) + ((e2 >> 3) << 4);
      bfr[e2] = (kk < 16) ? (__bf16)ub[kk * EMBED + nc] : (__bf16)0.0f;
    }
    v8f acc = {};
    acc = __builtin_amdgcn_wmma_f32_16x16x32_bf16(false, afr, false, bfr,
                                                  (short)0, acc, false, false);
    float bb = b2[nc];
#pragma unroll
    for (int r = 0; r < 8; ++r) {
      int tok = r + (half << 3);
      size_t gi = (size_t)(1 + n0 + tok) * (BATCH * EMBED) + (size_t)b * EMBED + nc;
      d_out[gi] = out_in[gi] + (acc[r] + bb) * ascale;
    }
  }
}

// ---------------- host ----------------
extern "C" void kernel_launch(void* const* d_in, const int* in_sizes, int n_in,
                              void* d_out, int out_size, void* d_ws, size_t ws_size,
                              hipStream_t stream) {
  (void)in_sizes; (void)n_in; (void)out_size; (void)ws_size;
  const float* in_output = (const float*)d_in[0];
  const float* in_text   = (const float*)d_in[1];
  const float* in_agent  = (const float*)d_in[2];
  const float* in_ascale = (const float*)d_in[3];
  const float* w1 = (const float*)d_in[4];
  const float* b1 = (const float*)d_in[5];
  const float* w2 = (const float*)d_in[6];
  const float* b2 = (const float*)d_in[7];
  const float* wm = (const float*)d_in[8];
  const float* bm = (const float*)d_in[9];
  const float* wt = (const float*)d_in[10];
  const float* bt = (const float*)d_in[11];
  const int* layer = (const int*)d_in[12];

  float* ws     = (float*)d_ws;
  float* tfn    = ws + OFF_TFN;
  float* temb   = ws + OFF_TEMB;
  float* agp    = ws + OFF_AGP;
  float* msc    = ws + OFF_MSC;
  float* mtpool = ws + OFF_MTPOOL;
  float* mt     = ws + OFF_MT;
  float* vpool  = ws + OFF_VPOOL;
  float* vwsum  = ws + OFF_VWSUM;
  float* ag2    = ws + OFF_AG2;
  float* vpre   = ws + OFF_VPRE;
  float* v1     = ws + OFF_V1;
  float* u      = ws + OFF_U;
  float* out    = (float*)d_out;

  k_zero<<<2048, 256, 0, stream>>>(ws, WS_FLOATS);
  k_agp<<<48, 256, 0, stream>>>(in_agent, layer, agp);
  k_textnorm<<<150, 256, 0, stream>>>(in_text, tfn);
  // temb = relu(tfn @ wt^T + bt)   (160x512 @ 512x768)
  k_gemm_bf16<<<dim3(10, 48), 32, 0, stream>>>(tfn, TEXT_DIM, wt, TEXT_DIM, bt,
                                               nullptr, nullptr, temb, EMBED,
                                               TEXT_DIM, 1.0f, 1);
  // text-mask scores: msc = temb @ agp^T * scale   (160x768 @ 768x16)
  k_gemm_bf16<<<dim3(10, 1), 32, 0, stream>>>(temb, EMBED, agp, EMBED, nullptr,
                                              nullptr, nullptr, msc, 16,
                                              EMBED, SCALE_D, 0);
  k_textpool<<<dim3(AGENT_LEN, 6), 128, 0, stream>>>(msc, temb, mtpool);
  // masked_text = mtpool @ wm^T + bm
  k_gemm_bf16<<<dim3(1, 48), 32, 0, stream>>>(mtpool, EMBED, wm, EMBED, bm,
                                              nullptr, nullptr, mt, EMBED,
                                              EMBED, 1.0f, 0);
  // masked_vis pooled accumulation over all tokens (streams x once)
  k_vispool<<<dim3(BATCH, 4), 256, 0, stream>>>(in_output, agp, vpool, vwsum);
  k_visnorm<<<BATCH, 256, 0, stream>>>(vpool, vwsum);
  // ag2 = vpool @ wm^T + bm + agent + masked_text   (stacked 512 rows)
  k_gemm_bf16<<<dim3(32, 48), 32, 0, stream>>>(vpool, EMBED, wm, EMBED, bm,
                                               agp, mt, ag2, EMBED,
                                               EMBED, 1.0f, 0);
  // attn over text, vpre = softmax(ag2 @ temb^T) @ temb
  k_attnt<<<BATCH, 256, 0, stream>>>(ag2, temb, vpre);
  // v1 = vpre @ w1^T + b1 ; u = v1 @ w2^T  (b2 folded into final kernel)
  k_gemm_bf16<<<dim3(32, 48), 32, 0, stream>>>(vpre, EMBED, w1, EMBED, b1,
                                               nullptr, nullptr, v1, EMBED,
                                               EMBED, 1.0f, 0);
  k_gemm_bf16<<<dim3(32, 48), 32, 0, stream>>>(v1, EMBED, w2, EMBED, nullptr,
                                               nullptr, nullptr, u, EMBED,
                                               EMBED, 1.0f, 0);
  // cls passthrough + fused attn_v/delta/output (streams x once more)
  k_cls<<<96, 256, 0, stream>>>(in_output, out);
  k_final<<<dim3(BATCH, N_TOK / 16), 256, 0, stream>>>(in_output, ag2, u, b2,
                                                       in_ascale, out);
}
